// HeteroGNN_80075370266805
// MI455X (gfx1250) — matrix operand order repaired and checked by reading, not compiled, not run on hardware
//
#include <hip/hip_runtime.h>
#include <hip/hip_bf16.h>
#include <math.h>

// ---------------------------------------------------------------------------
// HeteroGNN forward for MI455X (gfx1250, wave32, WMMA).
//   N=100000 nodes per type, E=600000 edges per relation, D=H=128, 2 layers.
// Dense GEMMs use v_wmma_f32_16x16x32_bf16 with W transposed in LDS so both
// A and B fragments load as contiguous ds_load_b128. Edge aggregation uses
// hardware float atomics. All kernels launched on `stream` (capture safe).
// ---------------------------------------------------------------------------

#define NN   100000
#define EE   600000
#define HD   128

typedef __attribute__((ext_vector_type(16))) __bf16 v16bf;
typedef __attribute__((ext_vector_type(8)))  float  v8f;

// ---------------------------------------------------------------------------
// GEMM: Out[nrows x 128] (+)= X[nrows x 128] @ W[128 x 128] (+ bias)
// Block = 256 threads = 8 waves; block owns 16 output rows, wave w owns
// columns [16w, 16w+16). K=128 swept as 4 WMMA steps of K=32 (bf16).
// ACCUM / HAS_BIAS are compile-time so the epilogue is branch-free.
// REQUIREMENT: nrows % 16 == 0 and grid.x == nrows/16 (exact tiling, no
// bounds checks anywhere).  NN = 100000 = 6250*16 satisfies this.
// ---------------------------------------------------------------------------
template <bool ACCUM, bool HAS_BIAS>
__global__ __launch_bounds__(256)
void hgnn_gemm128_bf16(const float* __restrict__ X,
                       const float* __restrict__ W,
                       const float* __restrict__ bias,
                       float* __restrict__ Out)
{
    __shared__ __bf16 sX [16][HD + 8];  // activation tile, [m][k]
    __shared__ __bf16 sWt[HD][HD + 8];  // weight TRANSPOSED, [n][k]

    const int tid  = threadIdx.x;
    const int wave = tid >> 5;          // 0..7
    const int lane = tid & 31;
    const int row0 = blockIdx.x * 16;

    // Stage W transposed: global read coalesced (i = k*128 + n), LDS write
    // scattered. sWt[n][k] = W[k,n] -> B fragments become contiguous in K.
    for (int i = tid; i < HD * HD; i += 256) {
        int k = i >> 7, n = i & 127;
        sWt[n][k] = (__bf16)W[i];
    }
    // Stage the 16x128 X tile with 128-bit global loads (4 floats/thread/iter).
    for (int i = tid; i < 16 * HD / 4; i += 256) {
        int r = i >> 5;                 // 32 float4 per row
        int c4 = (i & 31) * 4;
        float4 v = *(const float4*)(X + (size_t)(row0 + r) * HD + c4);
        sX[r][c4 + 0] = (__bf16)v.x;
        sX[r][c4 + 1] = (__bf16)v.y;
        sX[r][c4 + 2] = (__bf16)v.z;
        sX[r][c4 + 3] = (__bf16)v.w;
    }
    __syncthreads();

    const int half = lane >> 4;         // 0: lanes 0-15, 1: lanes 16-31
    const int mn   = lane & 15;         // M index for A, N index for B/C/D
    const int col0 = wave * 16;

    v8f acc = {};
    #pragma unroll
    for (int k0 = 0; k0 < HD; k0 += 32) {
        v16bf a, b;
        // A fragment: K order matches B. For a fixed lane the
        // 16 elements are one contiguous 16-run -> 2x ds_load_b128.
        //   j=0..15 : K = half*16 + j
        #pragma unroll
        for (int j = 0; j < 16; ++j) {
            a[j] = sX[mn][k0 + half * 16 + j];
        }
        // B fragment: 32x16, lanes 0-15 hold K=0..15, lanes 16-31 K=16..31.
        // With sWt[n][k] this is 16 contiguous bf16 -> 2x ds_load_b128.
        #pragma unroll
        for (int j = 0; j < 16; ++j) {
            b[j] = sWt[col0 + mn][k0 + half * 16 + j];
        }
        acc = __builtin_amdgcn_wmma_f32_16x16x32_bf16(
                  false, a, false, b, (short)0, acc, false, false);
    }

    // D layout: element r -> M = r + 8*half, N = lane&15.  Straight-line
    // epilogue: no bounds checks (grid tiles rows exactly), no dead FP adds.
    const int gcol = col0 + mn;
    float* orow = Out + (size_t)(row0 + half * 8) * HD + gcol;
    #pragma unroll
    for (int r = 0; r < 8; ++r) {
        float v = acc[r];
        if (HAS_BIAS) v += bias[gcol];
        if (ACCUM)    v += orow[(size_t)r * HD];
        orow[(size_t)r * HD] = v;
    }
}

// ---------------------------------------------------------------------------
// Row dot product: out[i] = dot(Hrow_i[0:128], avec[0:128]).  One wave/row.
// ---------------------------------------------------------------------------
__global__ __launch_bounds__(256)
void hgnn_rowdot(const float* __restrict__ Hm, const float* __restrict__ avec,
                 float* __restrict__ out, int nrows)
{
    int wave = threadIdx.x >> 5;
    int lane = threadIdx.x & 31;
    int row  = blockIdx.x * 8 + wave;
    if (row >= nrows) return;
    const float* h = Hm + (size_t)row * HD;
    float s = 0.0f;
    #pragma unroll
    for (int j = 0; j < 4; ++j) s += h[lane + 32 * j] * avec[lane + 32 * j];
    #pragma unroll
    for (int off = 16; off > 0; off >>= 1) s += __shfl_xor(s, off, 32);
    if (lane == 0) out[row] = s;
}

// ---------------------------------------------------------------------------
// Utility fill.
// ---------------------------------------------------------------------------
__global__ void hgnn_fill(float* __restrict__ p, float v, size_t n)
{
    size_t i = (size_t)blockIdx.x * blockDim.x + threadIdx.x;
    if (i < n) p[i] = v;
}

// sign-split float atomic max (init with -inf)
__device__ __forceinline__ void atomicMaxF(float* addr, float val)
{
    if (val >= 0.0f) atomicMax((int*)addr, __float_as_int(val));
    else             atomicMin((unsigned int*)addr, (unsigned int)__float_as_int(val));
}

// ---------------------------------------------------------------------------
// GAT attention pass 1: per-dst segment max of leaky_relu(as[src]+ad[dst]).
// ---------------------------------------------------------------------------
__global__ __launch_bounds__(256)
void hgnn_gat_max(const int* __restrict__ src, const int* __restrict__ dst,
                  const float* __restrict__ as_, const float* __restrict__ ad_,
                  float* __restrict__ amax, int ne)
{
    int e = blockIdx.x * blockDim.x + threadIdx.x;
    if (e >= ne) return;
    float a = as_[src[e]] + ad_[dst[e]];
    a = (a > 0.0f) ? a : 0.2f * a;
    atomicMaxF(amax + dst[e], a);
}

// ---------------------------------------------------------------------------
// GAT attention pass 2: ex[e] = exp(a - max[dst]); denom[dst] += ex.
// ---------------------------------------------------------------------------
__global__ __launch_bounds__(256)
void hgnn_gat_exp(const int* __restrict__ src, const int* __restrict__ dst,
                  const float* __restrict__ as_, const float* __restrict__ ad_,
                  const float* __restrict__ amax, float* __restrict__ ex,
                  float* __restrict__ denom, int ne)
{
    int e = blockIdx.x * blockDim.x + threadIdx.x;
    if (e >= ne) return;
    float a = as_[src[e]] + ad_[dst[e]];
    a = (a > 0.0f) ? a : 0.2f * a;
    float x = __expf(a - amax[dst[e]]);
    ex[e] = x;
    atomicAdd(denom + dst[e], x);
}

// ---------------------------------------------------------------------------
// GAT attention pass 3: out[dst] += (ex/denom[dst]) * hs[src].  1 wave/edge.
// ---------------------------------------------------------------------------
__global__ __launch_bounds__(256)
void hgnn_gat_scatter(const int* __restrict__ src, const int* __restrict__ dst,
                      const float* __restrict__ hs, const float* __restrict__ ex,
                      const float* __restrict__ denom, float* __restrict__ out,
                      int ne)
{
    int wave = threadIdx.x >> 5;
    int lane = threadIdx.x & 31;
    int e    = blockIdx.x * 8 + wave;
    if (e >= ne) return;
    int s = src[e], d = dst[e];
    float alpha = ex[e] / (denom[d] + 1e-16f);
    const float* hrow = hs  + (size_t)s * HD;
    float*       orow = out + (size_t)d * HD;
    #pragma unroll
    for (int j = 0; j < 4; ++j) {
        int c = lane + 32 * j;
        atomicAdd(orow + c, alpha * hrow[c]);
    }
}

// ---------------------------------------------------------------------------
// SAGE neighbor aggregation: agg[dst] += x[src]; cnt[dst] += 1.  1 wave/edge.
// ---------------------------------------------------------------------------
__global__ __launch_bounds__(256)
void hgnn_sage_agg(const int* __restrict__ src, const int* __restrict__ dst,
                   const float* __restrict__ x, float* __restrict__ agg,
                   float* __restrict__ cnt, int ne)
{
    int wave = threadIdx.x >> 5;
    int lane = threadIdx.x & 31;
    int e    = blockIdx.x * 8 + wave;
    if (e >= ne) return;
    int s = src[e], d = dst[e];
    const float* xr = x   + (size_t)s * HD;
    float*       ar = agg + (size_t)d * HD;
    #pragma unroll
    for (int j = 0; j < 4; ++j) {
        int c = lane + 32 * j;
        atomicAdd(ar + c, xr[c]);
    }
    if (lane == 0) atomicAdd(cnt + d, 1.0f);
}

// agg /= max(cnt, 1)
__global__ void hgnn_sage_mean(float* __restrict__ agg,
                               const float* __restrict__ cnt, int nrows)
{
    size_t i = (size_t)blockIdx.x * blockDim.x + threadIdx.x;
    if (i >= (size_t)nrows * HD) return;
    float c = cnt[i >> 7];
    c = (c > 1.0f) ? c : 1.0f;
    agg[i] *= (1.0f / c);
}

// out[n, :] += b[:]   (GAT bias, applied once per node)
__global__ void hgnn_add_bias(float* __restrict__ out,
                              const float* __restrict__ b, int nrows)
{
    size_t i = (size_t)blockIdx.x * blockDim.x + threadIdx.x;
    if (i >= (size_t)nrows * HD) return;
    out[i] += b[i & 127];
}

__global__ void hgnn_relu(float* __restrict__ p, size_t n)
{
    size_t i = (size_t)blockIdx.x * blockDim.x + threadIdx.x;
    if (i < n) p[i] = fmaxf(p[i], 0.0f);
}

// ---------------------------------------------------------------------------
// Host orchestration
// ---------------------------------------------------------------------------
extern "C" void kernel_launch(void* const* d_in, const int* in_sizes, int n_in,
                              void* d_out, int out_size, void* d_ws, size_t ws_size,
                              hipStream_t stream)
{
    (void)in_sizes; (void)n_in; (void)out_size; (void)ws_size;

    const size_t nf = (size_t)NN * HD;   // floats per node-feature buffer

    // Scratch carve-out from d_ws
    float* base = (float*)d_ws;
    float* buf[6];
    for (int i = 0; i < 6; ++i) buf[i] = base + (size_t)i * nf;
    float* cnt   = base + 6 * nf;
    float* amax  = cnt   + NN;
    float* denom = amax  + NN;
    float* as_   = denom + NN;
    float* ad_   = as_   + NN;
    float* ex    = ad_   + NN;           // EE floats

    auto P = [&](int i) { return (const float*)d_in[i]; };

    const int* ei_to  = (const int*)d_in[2];  // trained_on
    const int* ei_sim = (const int*)d_in[3];  // similar_to
    const int* ei_rev = (const int*)d_in[4];  // rev_trained_on
    const int* ei_tt  = (const int*)d_in[5];  // transfer_to

    const int GEMM_GRID = NN / 16;                // 6250 (exact tiling)
    const int ROW_GRID  = (NN + 7) / 8;           // 12500
    const int E_ELT     = (EE + 255) / 256;       // elementwise over edges
    const int E_WAVE    = (EE + 7) / 8;           // wave-per-edge
    const int NF_GRID   = (int)((nf + 255) / 256);

    auto fill = [&](float* p, float v, size_t n) {
        hgnn_fill<<<(int)((n + 255) / 256), 256, 0, stream>>>(p, v, n);
    };
    auto gemm_store = [&](const float* X, const float* W, float* Out) {
        hgnn_gemm128_bf16<false, false><<<GEMM_GRID, 256, 0, stream>>>(X, W, nullptr, Out);
    };
    auto gemm_acc = [&](const float* X, const float* W, float* Out) {
        hgnn_gemm128_bf16<true, false><<<GEMM_GRID, 256, 0, stream>>>(X, W, nullptr, Out);
    };
    auto gemm_acc_bias = [&](const float* X, const float* W, const float* bias,
                             float* Out) {
        hgnn_gemm128_bf16<true, true><<<GEMM_GRID, 256, 0, stream>>>(X, W, bias, Out);
    };

    // GAT relation: x_src @ Ws attended into acc (+ bias per node)
    auto run_gat = [&](const float* xsrc, const float* xdst, const int* ei,
                       const float* Ws, const float* Wd,
                       const float* a_s, const float* a_d,
                       const float* bias, float* acc) {
        const int* src = ei;
        const int* dst = ei + EE;
        float* hs = buf[4];
        float* hd = buf[5];
        gemm_store(xsrc, Ws, hs);
        gemm_store(xdst, Wd, hd);
        hgnn_rowdot<<<ROW_GRID, 256, 0, stream>>>(hs, a_s, as_, NN);
        hgnn_rowdot<<<ROW_GRID, 256, 0, stream>>>(hd, a_d, ad_, NN);
        fill(amax, -INFINITY, NN);
        fill(denom, 0.0f, NN);
        hgnn_gat_max<<<E_ELT, 256, 0, stream>>>(src, dst, as_, ad_, amax, EE);
        hgnn_gat_exp<<<E_ELT, 256, 0, stream>>>(src, dst, as_, ad_, amax, ex, denom, EE);
        hgnn_gat_scatter<<<E_WAVE, 256, 0, stream>>>(src, dst, hs, ex, denom, acc, EE);
        hgnn_add_bias<<<NF_GRID, 256, 0, stream>>>(acc, bias, NN);
    };

    // SAGE relation: mean_agg(x_src) @ Wl + bl + x_dst @ Wr  accumulated into acc
    auto run_sage = [&](const float* xsrc, const float* xdst, const int* ei,
                        const float* Wl, const float* Wr, const float* bl,
                        float* acc) {
        const int* src = ei;
        const int* dst = ei + EE;
        float* agg = buf[4];
        fill(agg, 0.0f, nf);
        fill(cnt, 0.0f, NN);
        hgnn_sage_agg<<<E_WAVE, 256, 0, stream>>>(src, dst, xsrc, agg, cnt, EE);
        hgnn_sage_mean<<<NF_GRID, 256, 0, stream>>>(agg, cnt, NN);
        gemm_acc_bias(agg, Wl, bl, acc);
        gemm_acc(xdst, Wr, acc);
    };

    const float* xm = (const float*)d_in[0];
    const float* xd = (const float*)d_in[1];
    float* dacc = nullptr;
    float* macc = nullptr;

    for (int l = 0; l < 2; ++l) {
        // Param leaf layout per layer (JAX pytree: dict keys sorted):
        //   rev_trained_on: Wl, Wr, bl          -> b+0..b+2
        //   similar_to:     Wl, Wr, bl          -> b+3..b+5
        //   trained_on:     Wd, Ws, a_d, a_s, b -> b+6..b+10
        //   transfer_to:    Wd, Ws, a_d, a_s, b -> b+11..b+15
        const int b = 6 + l * 16;
        dacc = (l == 0) ? buf[2] : buf[0];
        macc = (l == 0) ? buf[3] : buf[1];
        fill(dacc, 0.0f, nf);
        fill(macc, 0.0f, nf);

        // dataset destination: GAT(trained_on) + SAGE(similar_to) + GAT(transfer_to)
        run_gat (xm, xd, ei_to,  P(b + 7),  P(b + 6),  P(b + 9),  P(b + 8),  P(b + 10), dacc);
        run_sage(xd, xd, ei_sim, P(b + 3),  P(b + 4),  P(b + 5),  dacc);
        run_gat (xm, xd, ei_tt,  P(b + 12), P(b + 11), P(b + 14), P(b + 13), P(b + 15), dacc);
        // model destination: SAGE(rev_trained_on)
        run_sage(xd, xm, ei_rev, P(b + 0),  P(b + 1),  P(b + 2),  macc);

        hgnn_relu<<<NF_GRID, 256, 0, stream>>>(dacc, nf);
        hgnn_relu<<<NF_GRID, 256, 0, stream>>>(macc, nf);

        xm = macc;
        xd = dacc;
    }

    // Output: (x_model, x_dataset) concatenated.
    float* out = (float*)d_out;
    hipMemcpyAsync(out,      macc, nf * sizeof(float), hipMemcpyDeviceToDevice, stream);
    hipMemcpyAsync(out + nf, dacc, nf * sizeof(float), hipMemcpyDeviceToDevice, stream);
}
